// GQA_76725295776414
// MI455X (gfx1250) — compile-verified, matrix operand order checked by reading, not run on hardware
//
#include <hip/hip_runtime.h>
#include <hip/hip_bf16.h>
#include <cstdint>
#include <cstddef>

#define TPB 256

typedef __attribute__((ext_vector_type(16))) __bf16 v16bf;
typedef __attribute__((ext_vector_type(8)))  float  v8f;

namespace cfg {
constexpr int L = 2, D = 1024, HID = 4096, QH = 16, KVH = 4, V = 32000;
constexpr int B = 2, S = 2048, R = 20;
constexpr int HEAD = 64, N = B * S, KD = KVH * HEAD, G = QH / KVH;
constexpr float ALPHA = 32.0f;
}

__device__ __forceinline__ v8f v8f_zero() {
  v8f z;
#pragma unroll
  for (int i = 0; i < 8; ++i) z[i] = 0.0f;
  return z;
}

__device__ __forceinline__ void wait_async0() {
#if __has_builtin(__builtin_amdgcn_s_wait_asynccnt)
  __builtin_amdgcn_s_wait_asynccnt(0);
#else
  asm volatile("s_wait_asynccnt 0" ::: "memory");
#endif
}

// Async 16B global -> LDS copy (per-lane). ASYNCcnt-tracked.
__device__ __forceinline__ void async_copy_b128(unsigned lds_off, const void* gptr) {
  asm volatile("global_load_async_to_lds_b128 %0, %1, off"
               :: "v"(lds_off), "v"((unsigned long long)(size_t)gptr)
               : "memory");
}

// ---------------------------------------------------------------------------
// Embedding gather: x[n,d] = emb[xs[n], d]
// ---------------------------------------------------------------------------
__global__ __launch_bounds__(TPB) void embed_kernel(const int* __restrict__ xs,
                                                    const float* __restrict__ emb,
                                                    float* __restrict__ x) {
  using namespace cfg;
  size_t i = (size_t)blockIdx.x * TPB + threadIdx.x;
  if (i >= (size_t)N * D) return;
  int tok = (int)(i / D);
  int d   = (int)(i % D);
  x[i] = emb[(size_t)xs[tok] * D + d];
}

// ---------------------------------------------------------------------------
// RMSNorm: one block per row of D=1024; emits bf16 activations for WMMA GEMMs
// ---------------------------------------------------------------------------
__global__ __launch_bounds__(TPB) void rmsnorm_kernel(const float* __restrict__ x,
                                                      const float* __restrict__ sc,
                                                      __bf16* __restrict__ out) {
  using namespace cfg;
  const int row = blockIdx.x;
  const float* xr = x + (size_t)row * D;
  float s = 0.0f;
  for (int i = threadIdx.x; i < D; i += TPB) { float v = xr[i]; s += v * v; }
#pragma unroll
  for (int off = 1; off < 32; off <<= 1) s += __shfl_xor(s, off, 32);
  __shared__ float red[8];
  if ((threadIdx.x & 31) == 0) red[threadIdx.x >> 5] = s;
  __syncthreads();
  float tot = 0.0f;
#pragma unroll
  for (int w = 0; w < 8; ++w) tot += red[w];
  const float inv = rsqrtf(tot / (float)D + 1e-5f);
  __bf16* orow = out + (size_t)row * D;
  for (int i = threadIdx.x; i < D; i += TPB) orow[i] = (__bf16)(sc[i] * xr[i] * inv);
}

// ---------------------------------------------------------------------------
// Weight convert+transpose: Wt[m,k] (bf16) = W[k,m] (f32); rows contiguous in K
// ---------------------------------------------------------------------------
__global__ __launch_bounds__(TPB) void wcvt_t_kernel(const float* __restrict__ W,
                                                     __bf16* __restrict__ Wt,
                                                     int K, int M) {
  size_t i = (size_t)blockIdx.x * TPB + threadIdx.x;
  if (i >= (size_t)K * M) return;
  int m = (int)(i / K);
  int k2 = (int)(i % K);
  Wt[i] = (__bf16)W[(size_t)k2 * M + m];
}

// ---------------------------------------------------------------------------
// RoPE (f32 in, bf16 out). One thread per (token, head, pair).
// ---------------------------------------------------------------------------
__global__ __launch_bounds__(TPB) void rope_kernel(const float* __restrict__ in,
                                                   __bf16* __restrict__ out, int NH) {
  using namespace cfg;
  size_t i = (size_t)blockIdx.x * TPB + threadIdx.x;
  size_t tot = (size_t)N * NH * (HEAD / 2);
  if (i >= tot) return;
  int pr = (int)(i % (HEAD / 2));
  size_t th = i / (HEAD / 2);
  int head = (int)(th % NH);
  int tokidx = (int)(th / NH);
  int pos = tokidx % S;  // startpos == 0
  // freq = THETA^(-2*pr/HEAD); ln(50000) = 10.819778284410283
  float freq = __expf(-(2.0f * (float)pr / (float)HEAD) * 10.8197782844f);
  float sn, c;
  __sincosf((float)pos * freq, &sn, &c);
  const float* p = in + ((size_t)tokidx * NH + head) * HEAD + 2 * pr;
  float t0 = p[0], t1 = p[1];
  __bf16* q = out + ((size_t)tokidx * NH + head) * HEAD + 2 * pr;
  q[0] = (__bf16)(t0 * c - t1 * sn);
  q[1] = (__bf16)(t0 * sn + t1 * c);
}

__global__ __launch_bounds__(TPB) void cvt_bf16_kernel(const float* __restrict__ in,
                                                       __bf16* __restrict__ out, size_t n) {
  size_t i = (size_t)blockIdx.x * TPB + threadIdx.x;
  if (i < n) out[i] = (__bf16)in[i];
}

// m = g1 * silu(g2), emitted as bf16 for the W3 WMMA GEMM
__global__ __launch_bounds__(TPB) void mulsilu_kernel(const float* __restrict__ g1,
                                                      const float* __restrict__ g2,
                                                      __bf16* __restrict__ m, size_t n) {
  size_t i = (size_t)blockIdx.x * TPB + threadIdx.x;
  if (i < n) {
    float b = g2[i];
    m[i] = (__bf16)(g1[i] * (b / (1.0f + __expf(-b))));
  }
}

// ---------------------------------------------------------------------------
// LoRA intermediate: T[n,r] = h[n,:] @ Aw[:,r] + ab[r]   (R = 20, tiny)
// ---------------------------------------------------------------------------
__global__ __launch_bounds__(TPB) void lorat_kernel(const __bf16* __restrict__ h,
                                                    const float* __restrict__ Aw,
                                                    const float* __restrict__ ab,
                                                    float* __restrict__ T, int K) {
  using namespace cfg;
  int i = blockIdx.x * TPB + threadIdx.x;
  if (i >= N * R) return;
  int row = i / R, r = i % R;
  const __bf16* hr = h + (size_t)row * K;
  float s = ab[r];
  for (int k2 = 0; k2 < K; ++k2) s += (float)hr[k2] * Aw[(size_t)k2 * R + r];
  T[i] = s;
}

// ---------------------------------------------------------------------------
// Tiled bf16 WMMA GEMM with async-LDS staging.
//   A  : bf16 [N,K] row-major (lda == K)
//   Wt : bf16 [M,K] row-major (pre-transposed weight; row = output column)
//   C  : f32 [N,M] = A @ Wt^T + bias (+ (T@Bl + bb)/ALPHA) (+ Resid)
// Block: 128x128 tile, BK=32, 8 waves (2x4); wave: 4x2 16x16 WMMA tiles.
// Tiles staged into LDS with global_load_async_to_lds_b128 (ASYNCcnt).
// ---------------------------------------------------------------------------
template <int LORA, int RESID>
__global__ __launch_bounds__(TPB) void gemm_kernel(
    const __bf16* __restrict__ A,
    const __bf16* __restrict__ Wt,
    const float* __restrict__ bias,
    const float* __restrict__ T,
    const float* __restrict__ Bl,
    const float* __restrict__ bb,
    const float* __restrict__ Resid,
    float* __restrict__ C, int ldc,
    int K, int Mcols) {
  constexpr int BM = 128, BN = 128, BK = 32;
  constexpr int LSE = 40;          // padded LDS row stride (bf16 elems) = 80 B
  __shared__ __attribute__((aligned(16))) __bf16 As[BM * LSE];
  __shared__ __attribute__((aligned(16))) __bf16 Bs[BN * LSE];

  const int tid = threadIdx.x;
  const int wave = tid >> 5, lane = tid & 31;
  const int wm = wave >> 2, wn = wave & 3;  // 2x4 wave grid
  const int half = lane >> 4, n16 = lane & 15;
  const int k0 = half * 8;
  const int rowBase = blockIdx.y * BM;
  const int colBase = blockIdx.x * BN;
  const unsigned asBase = (unsigned)(size_t)&As[0];
  const unsigned bsBase = (unsigned)(size_t)&Bs[0];

  v8f acc[4][2];
#pragma unroll
  for (int ti = 0; ti < 4; ++ti)
#pragma unroll
    for (int tj = 0; tj < 2; ++tj) acc[ti][tj] = v8f_zero();

  for (int kk = 0; kk < K; kk += BK) {
    // ---- async stage: 128 rows x 64 B per tile = 512 x b128 chunks each
#pragma unroll
    for (int i = 0; i < 2; ++i) {
      int idx = tid + i * TPB;      // 0..511
      int r = idx >> 2;             // tile row
      int c = idx & 3;              // 16B chunk within row
      async_copy_b128(asBase + (unsigned)(r * LSE * 2 + c * 16),
                      A + (size_t)(rowBase + r) * K + kk + c * 8);
    }
#pragma unroll
    for (int i = 0; i < 2; ++i) {
      int idx = tid + i * TPB;
      int r = idx >> 2;
      int c = idx & 3;
      async_copy_b128(bsBase + (unsigned)(r * LSE * 2 + c * 16),
                      Wt + (size_t)(colBase + r) * K + kk + c * 8);
    }
    wait_async0();
    __syncthreads();

    // ---- WMMA: one K=32 step consumes the whole LDS tile
    v16bf af[4], bfr[2];
#pragma unroll
    for (int ti = 0; ti < 4; ++ti) {
      const __bf16* p = &As[(wm * 64 + ti * 16 + n16) * LSE + k0];
      uint4* d = reinterpret_cast<uint4*>(&af[ti]);
      d[0] = *reinterpret_cast<const uint4*>(p);
      d[1] = *reinterpret_cast<const uint4*>(p + 16);
    }
#pragma unroll
    for (int tj = 0; tj < 2; ++tj) {
      const __bf16* p = &Bs[(wn * 32 + tj * 16 + n16) * LSE + k0];
      uint4* d = reinterpret_cast<uint4*>(&bfr[tj]);
      d[0] = *reinterpret_cast<const uint4*>(p);
      d[1] = *reinterpret_cast<const uint4*>(p + 16);
    }
#pragma unroll
    for (int ti = 0; ti < 4; ++ti)
#pragma unroll
      for (int tj = 0; tj < 2; ++tj)
        acc[ti][tj] = __builtin_amdgcn_wmma_f32_16x16x32_bf16(
            false, af[ti], false, bfr[tj], (short)0, acc[ti][tj], false, false);
    __syncthreads();
  }

  // ---- epilogue: bias + LoRA + residual, f32 store
#pragma unroll
  for (int ti = 0; ti < 4; ++ti) {
#pragma unroll
    for (int tj = 0; tj < 2; ++tj) {
      const int col = colBase + wn * 32 + tj * 16 + n16;
      const float bval = bias ? bias[col] : 0.0f;
#pragma unroll
      for (int r = 0; r < 8; ++r) {
        const int row = rowBase + wm * 64 + ti * 16 + half * 8 + r;
        float vv = acc[ti][tj][r] + bval;
        if (LORA) {
          float s = bb[col];
          const float* tr = T + (size_t)row * cfg::R;
#pragma unroll
          for (int rr = 0; rr < cfg::R; ++rr)
            s += tr[rr] * Bl[(size_t)rr * Mcols + col];
          vv += s * (1.0f / cfg::ALPHA);
        }
        if (RESID) vv += Resid[(size_t)row * ldc + col];
        C[(size_t)row * ldc + col] = vv;
      }
    }
  }
}

// ---------------------------------------------------------------------------
// Flash attention (causal, GQA). One wave = 16 queries x full HEAD=64.
// Block = 8 waves = 128 queries of one (batch, q-head).
// Key blocks of 32: QK^T via 4 WMMAs, online softmax, P@V via 4 WMMAs.
// q/k/v bf16: q [B,S,QH,64], k/v [B,S,KVH,64]; o bf16 [B,S,QH,64] (= [N,D]).
// ---------------------------------------------------------------------------
__global__ __launch_bounds__(TPB) void attn_fa_kernel(const __bf16* __restrict__ q,
                                                      const __bf16* __restrict__ k,
                                                      const __bf16* __restrict__ v,
                                                      __bf16* __restrict__ o) {
  using namespace cfg;
  __shared__ __attribute__((aligned(16))) __bf16 Pl[8][16][40];
  const int tid = threadIdx.x, wave = tid >> 5, lane = tid & 31;
  const int half = lane >> 4, m = lane & 15, k0 = half * 8;
  const int bh = blockIdx.x;
  const int b = bh / QH, hq = bh % QH, hk = hq / G;
  const int qBase = blockIdx.y * 128 + wave * 16;
  const float scale = 0.125f;  // 1/sqrt(64)

  v16bf qf[2];
  {
    const __bf16* qrow = q + (((size_t)(b * S + qBase + m)) * QH + hq) * HEAD;
#pragma unroll
    for (int f = 0; f < 2; ++f) {
      uint4* d = reinterpret_cast<uint4*>(&qf[f]);
      d[0] = *reinterpret_cast<const uint4*>(qrow + f * 32 + k0);
      d[1] = *reinterpret_cast<const uint4*>(qrow + f * 32 + k0 + 16);
    }
  }

  v8f oacc[4];
#pragma unroll
  for (int dt = 0; dt < 4; ++dt) oacc[dt] = v8f_zero();
  float mi[8], li[8];
#pragma unroll
  for (int r = 0; r < 8; ++r) { mi[r] = -3.0e38f; li[r] = 0.0f; }

  const int kmax = qBase + 16;  // causal
  for (int kb0 = 0; kb0 < kmax; kb0 += 32) {
    v8f sc[2];
#pragma unroll
    for (int nt = 0; nt < 2; ++nt) {
      sc[nt] = v8f_zero();
      int key = kb0 + nt * 16 + m;
      if (key > S - 1) key = S - 1;  // clamped loads masked below
      const __bf16* krow = k + (((size_t)(b * S + key)) * KVH + hk) * HEAD;
      v16bf kf[2];
#pragma unroll
      for (int f = 0; f < 2; ++f) {
        uint4* d = reinterpret_cast<uint4*>(&kf[f]);
        d[0] = *reinterpret_cast<const uint4*>(krow + f * 32 + k0);
        d[1] = *reinterpret_cast<const uint4*>(krow + f * 32 + k0 + 16);
      }
      sc[nt] = __builtin_amdgcn_wmma_f32_16x16x32_bf16(false, qf[0], false, kf[0],
                                                       (short)0, sc[nt], false, false);
      sc[nt] = __builtin_amdgcn_wmma_f32_16x16x32_bf16(false, qf[1], false, kf[1],
                                                       (short)0, sc[nt], false, false);
    }
#pragma unroll
    for (int r = 0; r < 8; ++r) {
      const int qi = qBase + half * 8 + r;
      float s0 = sc[0][r] * scale;
      float s1 = sc[1][r] * scale;
      if (kb0 + m > qi) s0 = -3.0e38f;
      if (kb0 + 16 + m > qi) s1 = -3.0e38f;
      float rm = fmaxf(s0, s1);
#pragma unroll
      for (int off = 1; off < 16; off <<= 1) rm = fmaxf(rm, __shfl_xor(rm, off, 32));
      const float mnew = fmaxf(mi[r], rm);
      const float corr = __expf(mi[r] - mnew);
      const float p0 = __expf(s0 - mnew);
      const float p1 = __expf(s1 - mnew);
      float ls = p0 + p1;
#pragma unroll
      for (int off = 1; off < 16; off <<= 1) ls += __shfl_xor(ls, off, 32);
      li[r] = li[r] * corr + ls;
      mi[r] = mnew;
#pragma unroll
      for (int dt = 0; dt < 4; ++dt) oacc[dt][r] *= corr;
      Pl[wave][half * 8 + r][m] = (__bf16)p0;
      Pl[wave][half * 8 + r][16 + m] = (__bf16)p1;
    }
#if __has_builtin(__builtin_amdgcn_s_wait_dscnt)
    __builtin_amdgcn_s_wait_dscnt(0);
#else
    asm volatile("s_wait_dscnt 0" ::: "memory");
#endif
    v16bf pf;
    {
      const __bf16* pp = &Pl[wave][m][k0];
      uint4* d = reinterpret_cast<uint4*>(&pf);
      d[0] = *reinterpret_cast<const uint4*>(pp);
      d[1] = *reinterpret_cast<const uint4*>(pp + 16);
    }
#pragma unroll
    for (int dt = 0; dt < 4; ++dt) {
      v16bf vf;
      __bf16* vp = reinterpret_cast<__bf16*>(&vf);
#pragma unroll
      for (int i = 0; i < 8; ++i) {
        int key0 = kb0 + k0 + i;      if (key0 > S - 1) key0 = S - 1;
        int key1 = kb0 + k0 + 16 + i; if (key1 > S - 1) key1 = S - 1;
        vp[i]     = v[(((size_t)(b * S + key0)) * KVH + hk) * HEAD + dt * 16 + m];
        vp[8 + i] = v[(((size_t)(b * S + key1)) * KVH + hk) * HEAD + dt * 16 + m];
      }
      oacc[dt] = __builtin_amdgcn_wmma_f32_16x16x32_bf16(false, pf, false, vf,
                                                         (short)0, oacc[dt], false, false);
    }
  }
#pragma unroll
  for (int r = 0; r < 8; ++r) {
    const int qi = qBase + half * 8 + r;
    const float inv = 1.0f / li[r];
    __bf16* orow = o + (((size_t)(b * S + qi)) * QH + hq) * HEAD;
#pragma unroll
    for (int dt = 0; dt < 4; ++dt) orow[dt * 16 + m] = (__bf16)(oacc[dt][r] * inv);
  }
}

// ---------------------------------------------------------------------------
// Orchestration
// ---------------------------------------------------------------------------
extern "C" void kernel_launch(void* const* d_in, const int* in_sizes, int n_in,
                              void* d_out, int out_size, void* d_ws, size_t ws_size,
                              hipStream_t stream) {
  using namespace cfg;
  const int*   xs          = (const int*)  d_in[0];
  const float* emb         = (const float*)d_in[1];
  const float* Wq          = (const float*)d_in[2];
  const float* bq          = (const float*)d_in[3];
  const float* Wk          = (const float*)d_in[4];
  const float* bk          = (const float*)d_in[5];
  const float* Wv          = (const float*)d_in[6];
  const float* bv          = (const float*)d_in[7];
  const float* Wo          = (const float*)d_in[8];
  const float* bo          = (const float*)d_in[9];
  const float* att_scale   = (const float*)d_in[10];
  const float* mlp_scale   = (const float*)d_in[11];
  const float* final_scale = (const float*)d_in[12];
  const float* W1          = (const float*)d_in[13];
  const float* A1          = (const float*)d_in[14];
  const float* a1b         = (const float*)d_in[15];
  const float* B1          = (const float*)d_in[16];
  const float* b1b         = (const float*)d_in[17];
  const float* W2          = (const float*)d_in[18];
  const float* A2          = (const float*)d_in[19];
  const float* a2b         = (const float*)d_in[20];
  const float* B2          = (const float*)d_in[21];
  const float* b2b         = (const float*)d_in[22];
  const float* W3          = (const float*)d_in[23];
  const float* A3          = (const float*)d_in[24];
  const float* a3b         = (const float*)d_in[25];
  const float* B3          = (const float*)d_in[26];
  const float* b3b         = (const float*)d_in[27];
  (void)in_sizes; (void)n_in; (void)out_size;  // startpos assumed 0 (d_in[28])

  // Workspace carve (~237 MB required)
  char* ws = (char*)d_ws;
  size_t off = 0;
  auto carve = [&](size_t bytes) -> char* {
    char* p = ws + off;
    off += (bytes + 255) & ~(size_t)255;
    return p;
  };
  float*  x  = (float*) carve((size_t)N * D * 4);
  __bf16* hb = (__bf16*)carve((size_t)N * D * 2);
  float*  qf = (float*) carve((size_t)N * D * 4);
  float*  kf = (float*) carve((size_t)N * KD * 4);
  float*  vf = (float*) carve((size_t)N * KD * 4);
  __bf16* qb = (__bf16*)carve((size_t)N * D * 2);
  __bf16* kb = (__bf16*)carve((size_t)N * KD * 2);
  __bf16* vb = (__bf16*)carve((size_t)N * KD * 2);
  __bf16* ob = (__bf16*)carve((size_t)N * D * 2);
  float*  g1 = (float*) carve((size_t)N * HID * 4);
  float*  g2 = (float*) carve((size_t)N * HID * 4);  // reused as bf16 emb copy
  __bf16* mb = (__bf16*)carve((size_t)N * HID * 2);
  float*  T  = (float*) carve((size_t)N * R * 4);
  __bf16* wt = (__bf16*)carve((size_t)HID * D * 2);  // largest transposed weight
  (void)ws_size;

  const dim3 blk(TPB);
  auto grid1 = [](size_t n) { return dim3((unsigned)((n + TPB - 1) / TPB)); };

  embed_kernel<<<grid1((size_t)N * D), blk, 0, stream>>>(xs, emb, x);

  for (int l = 0; l < L; ++l) {
    const float* Wq_l = Wq + (size_t)l * D * D;
    const float* bq_l = bq + (size_t)l * D;
    const float* Wk_l = Wk + (size_t)l * D * KD;
    const float* bk_l = bk + (size_t)l * KD;
    const float* Wv_l = Wv + (size_t)l * D * KD;
    const float* bv_l = bv + (size_t)l * KD;
    const float* Wo_l = Wo + (size_t)l * D * D;
    const float* bo_l = bo + (size_t)l * D;
    const float* W1_l = W1 + (size_t)l * D * HID;
    const float* A1_l = A1 + (size_t)l * D * R;
    const float* a1b_l = a1b + (size_t)l * R;
    const float* B1_l = B1 + (size_t)l * R * HID;
    const float* b1b_l = b1b + (size_t)l * HID;
    const float* W2_l = W2 + (size_t)l * D * HID;
    const float* A2_l = A2 + (size_t)l * D * R;
    const float* a2b_l = a2b + (size_t)l * R;
    const float* B2_l = B2 + (size_t)l * R * HID;
    const float* b2b_l = b2b + (size_t)l * HID;
    const float* W3_l = W3 + (size_t)l * HID * D;
    const float* A3_l = A3 + (size_t)l * HID * R;
    const float* a3b_l = a3b + (size_t)l * R;
    const float* B3_l = B3 + (size_t)l * R * D;
    const float* b3b_l = b3b + (size_t)l * D;

    // --- attention block
    rmsnorm_kernel<<<N, blk, 0, stream>>>(x, att_scale + (size_t)l * D, hb);
    wcvt_t_kernel<<<grid1((size_t)D * D), blk, 0, stream>>>(Wq_l, wt, D, D);
    gemm_kernel<0, 0><<<dim3(D / 128, N / 128), blk, 0, stream>>>(
        hb, wt, bq_l, nullptr, nullptr, nullptr, nullptr, qf, D, D, D);
    wcvt_t_kernel<<<grid1((size_t)D * KD), blk, 0, stream>>>(Wk_l, wt, D, KD);
    gemm_kernel<0, 0><<<dim3(KD / 128, N / 128), blk, 0, stream>>>(
        hb, wt, bk_l, nullptr, nullptr, nullptr, nullptr, kf, KD, D, KD);
    wcvt_t_kernel<<<grid1((size_t)D * KD), blk, 0, stream>>>(Wv_l, wt, D, KD);
    gemm_kernel<0, 0><<<dim3(KD / 128, N / 128), blk, 0, stream>>>(
        hb, wt, bv_l, nullptr, nullptr, nullptr, nullptr, vf, KD, D, KD);
    rope_kernel<<<grid1((size_t)N * QH * (HEAD / 2)), blk, 0, stream>>>(qf, qb, QH);
    rope_kernel<<<grid1((size_t)N * KVH * (HEAD / 2)), blk, 0, stream>>>(kf, kb, KVH);
    cvt_bf16_kernel<<<grid1((size_t)N * KD), blk, 0, stream>>>(vf, vb, (size_t)N * KD);
    attn_fa_kernel<<<dim3(B * QH, S / 128), blk, 0, stream>>>(qb, kb, vb, ob);
    wcvt_t_kernel<<<grid1((size_t)D * D), blk, 0, stream>>>(Wo_l, wt, D, D);
    gemm_kernel<0, 1><<<dim3(D / 128, N / 128), blk, 0, stream>>>(
        ob, wt, bo_l, nullptr, nullptr, nullptr, x, x, D, D, D);

    // --- MLP block with LoRA epilogues
    rmsnorm_kernel<<<N, blk, 0, stream>>>(x, mlp_scale + (size_t)l * D, hb);
    lorat_kernel<<<grid1((size_t)N * R), blk, 0, stream>>>(hb, A1_l, a1b_l, T, D);
    wcvt_t_kernel<<<grid1((size_t)D * HID), blk, 0, stream>>>(W1_l, wt, D, HID);
    gemm_kernel<1, 0><<<dim3(HID / 128, N / 128), blk, 0, stream>>>(
        hb, wt, nullptr, T, B1_l, b1b_l, nullptr, g1, HID, D, HID);
    lorat_kernel<<<grid1((size_t)N * R), blk, 0, stream>>>(hb, A2_l, a2b_l, T, D);
    wcvt_t_kernel<<<grid1((size_t)D * HID), blk, 0, stream>>>(W2_l, wt, D, HID);
    gemm_kernel<1, 0><<<dim3(HID / 128, N / 128), blk, 0, stream>>>(
        hb, wt, nullptr, T, B2_l, b2b_l, nullptr, g2, HID, D, HID);
    mulsilu_kernel<<<grid1((size_t)N * HID), blk, 0, stream>>>(g1, g2, mb, (size_t)N * HID);
    lorat_kernel<<<grid1((size_t)N * R), blk, 0, stream>>>(mb, A3_l, a3b_l, T, HID);
    wcvt_t_kernel<<<grid1((size_t)HID * D), blk, 0, stream>>>(W3_l, wt, HID, D);
    gemm_kernel<1, 1><<<dim3(D / 128, N / 128), blk, 0, stream>>>(
        mb, wt, nullptr, T, B3_l, b3b_l, x, x, D, HID, D);
  }

  // --- final norm + logits (f @ emb^T); emb is already [V][D] = [M][K]
  rmsnorm_kernel<<<N, blk, 0, stream>>>(x, final_scale, hb);
  __bf16* embb = (__bf16*)g2;  // 62.5 MB fits in the 64 MB g2 slot
  cvt_bf16_kernel<<<grid1((size_t)V * D), blk, 0, stream>>>(emb, embb, (size_t)V * D);
  gemm_kernel<0, 0><<<dim3(V / 128, N / 128), blk, 0, stream>>>(
      hb, embb, nullptr, nullptr, nullptr, nullptr, nullptr,
      (float*)d_out, V, D, V);
}
